// GNNProcessor_25451976196263
// MI455X (gfx1250) — compile-verified, matrix operand order checked by reading, not run on hardware
//
#include <hip/hip_runtime.h>
#include <hip/hip_bf16.h>
#include <math.h>

typedef __attribute__((ext_vector_type(16))) _Float16 v16h;
typedef __attribute__((ext_vector_type(8)))  float    v8f;

#define LDIM 6
#define HID  16
#define DIN  24

// ---------------------------------------------------------------------------
// Utility: zero a float region of workspace
// ---------------------------------------------------------------------------
__global__ void zero_ws_kernel(float* __restrict__ w, int n) {
    int i = blockIdx.x * blockDim.x + threadIdx.x;
    if (i < n) w[i] = 0.0f;
}

// ---------------------------------------------------------------------------
// Degree: deg[dst] += 1 per edge
// ---------------------------------------------------------------------------
__global__ void degree_kernel(const int* __restrict__ dstI, float* __restrict__ deg, int E_) {
    int e = blockIdx.x * blockDim.x + threadIdx.x;
    if (e < E_) atomicAdd(&deg[dstI[e]], 1.0f);
}

// ---------------------------------------------------------------------------
// Edge MLP layer: one wave32 handles 16-edge tiles, grid-stride over tiles.
//   GEMM1: (16 x 24pad32) @ (24pad32 x 16)    via V_WMMA_F32_16X16X32_F16
//   GEMM2: (16 x 16pad32) @ (16pad32 x 6pad16) via V_WMMA_F32_16X16X32_F16
// m_in layout per edge (K index):
//   [0..5] X[src]  [6..11] X[dst]  [12..15] x[src]  [16..19] x[dst]  [20..23] edge_attr
// A layout (16-bit, 16x32): lanes 0-15 hold K{0-7,16-23}; lanes 16-31 hold K{8-15,24-31}.
// B layout mirrors A with N = lane&15. C layout: VGPR v -> M = v + (hi?8:0), N = lane&15.
// B fragments + biases are wave-invariant across tiles -> hoisted out of the loop.
// ---------------------------------------------------------------------------
__global__ __launch_bounds__(256) void edge_mlp_kernel(
    const float* __restrict__ X, const float* __restrict__ xnode,
    const float* __restrict__ eattr,
    const int* __restrict__ srcI, const int* __restrict__ dstI,
    const float* __restrict__ W1, const float* __restrict__ b1,
    const float* __restrict__ W2, const float* __restrict__ b2,
    float* __restrict__ Xacc, int E_) {

    __shared__ float lds_h[8][16][17];   // [wave][edge-in-tile][hid], padded stride

    const int lane = threadIdx.x & 31;
    const int wave = threadIdx.x >> 5;
    const int row  = lane & 15;          // M (edge within tile) for A; N (col) for B/C
    const bool hi  = lane >= 16;
    const int n    = row;                // output column this lane owns in C
    const int kb   = hi ? 8 : 0;
    const int mb   = hi ? 8 : 0;

    // ---- wave-invariant: B1 = W1 (24x16 padded to 32x16) ----
    v16h bw;
    if (!hi) {
        #pragma unroll
        for (int i = 0; i < 8; ++i) bw[i]     = (_Float16)W1[i * HID + n];         // K0..7
        #pragma unroll
        for (int i = 0; i < 8; ++i) bw[8 + i] = (_Float16)W1[(16 + i) * HID + n];  // K16..23
    } else {
        #pragma unroll
        for (int i = 0; i < 8; ++i) bw[i]     = (_Float16)W1[(8 + i) * HID + n];   // K8..15
        #pragma unroll
        for (int i = 0; i < 8; ++i) bw[8 + i] = (_Float16)0.0f;                    // K24..31
    }

    // ---- wave-invariant: B2 = W2 (16x6 padded to 32x16), branch-free ----
    const bool useCol = (n < LDIM);
    const int  nc     = useCol ? n : 0;       // clamped column (load always in-bounds)
    v16h bw2;
    #pragma unroll
    for (int i = 0; i < 8; ++i) {
        float wv = W2[(kb + i) * LDIM + nc];  // unconditional load, K(kb..kb+7)
        bw2[i] = useCol ? (_Float16)wv : (_Float16)0.0f;
    }
    #pragma unroll
    for (int i = 0; i < 8; ++i) bw2[8 + i] = (_Float16)0.0f;   // K16..31 (and 24..31) pad

    const float bias1 = b1[n];
    const float bias2 = b2[nc];

    const int numTiles   = (E_ + 15) >> 4;
    const int wavesTotal = gridDim.x * (blockDim.x >> 5);
    const int waveGlobal = blockIdx.x * (blockDim.x >> 5) + wave;

    for (int tile = waveGlobal; tile < numTiles; tile += wavesTotal) {
        const int tileBase = tile << 4;
        int e  = tileBase + row;
        int ec = (e < E_) ? e : (E_ - 1);     // clamp; invalid rows never written back
        int s  = srcI[ec];
        int d  = dstI[ec];

        // stream hint for the edge-attr stream (global_prefetch_b8)
        __builtin_prefetch(&eattr[(size_t)ec * 4 + 2048], 0, 1);

        // ---- build A (m_in tile, f32 -> f16) ----
        v16h a;
        if (!hi) {
            #pragma unroll
            for (int i = 0; i < 6; ++i) a[i] = (_Float16)X[(size_t)s * LDIM + i];      // K0..5
            a[6] = (_Float16)X[(size_t)d * LDIM + 0];                                  // K6
            a[7] = (_Float16)X[(size_t)d * LDIM + 1];                                  // K7
            #pragma unroll
            for (int i = 0; i < 4; ++i) a[8 + i]  = (_Float16)xnode[(size_t)d * 4 + i];   // K16..19
            #pragma unroll
            for (int i = 0; i < 4; ++i) a[12 + i] = (_Float16)eattr[(size_t)ec * 4 + i];  // K20..23
        } else {
            #pragma unroll
            for (int i = 0; i < 4; ++i) a[i]     = (_Float16)X[(size_t)d * LDIM + 2 + i]; // K8..11
            #pragma unroll
            for (int i = 0; i < 4; ++i) a[4 + i] = (_Float16)xnode[(size_t)s * 4 + i];    // K12..15
            #pragma unroll
            for (int i = 0; i < 8; ++i) a[8 + i] = (_Float16)0.0f;                        // K24..31
        }

        v8f c = {};
        c = __builtin_amdgcn_wmma_f32_16x16x32_f16(false, a, false, bw, (short)0, c, false, false);

        // ---- bias + ReLU, restage through LDS (C layout -> A layout transpose) ----
        #pragma unroll
        for (int v = 0; v < 8; ++v) {
            lds_h[wave][v + mb][n] = fmaxf(c[v] + bias1, 0.0f);
        }
        // wave-local fence: LDS ops from one wave are processed in order; just make
        // sure the compiler doesn't reorder and the DS stores are retired.
        __builtin_amdgcn_wave_barrier();
        asm volatile("s_wait_dscnt 0x0" ::: "memory");

        // ---- A2 = hidden tile (16x16 padded to 16x32) ----
        v16h a2;
        #pragma unroll
        for (int i = 0; i < 8; ++i) a2[i] = (_Float16)lds_h[wave][row][kb + i];
        #pragma unroll
        for (int i = 0; i < 8; ++i) a2[8 + i] = (_Float16)0.0f;

        v8f c2 = {};
        c2 = __builtin_amdgcn_wmma_f32_16x16x32_f16(false, a2, false, bw2, (short)0, c2, false, false);

        // ---- broadcast dst indices across the wave (no global reload) ----
        int dd[8];
        #pragma unroll
        for (int v = 0; v < 8; ++v) dd[v] = __shfl(d, v + mb, 32);

        // ---- scatter messages: Xacc[dst] += m  (segment sum) ----
        if (useCol) {
            #pragma unroll
            for (int v = 0; v < 8; ++v) {
                int e2 = tileBase + v + mb;
                if (e2 < E_) {
                    atomicAdd(&Xacc[(size_t)dd[v] * LDIM + n], c2[v] + bias2);
                }
            }
        }
    }
}

// ---------------------------------------------------------------------------
// Node update: X = relu(Xacc / max(deg,1)); re-zero accumulator
// ---------------------------------------------------------------------------
__global__ void node_update_kernel(float* __restrict__ Xacc, const float* __restrict__ deg,
                                   float* __restrict__ Xcur, int N_) {
    int i = blockIdx.x * blockDim.x + threadIdx.x;
    if (i < N_ * LDIM) {
        float dg = fmaxf(deg[i / LDIM], 1.0f);
        float v  = Xacc[i] / dg;
        Xcur[i]  = fmaxf(v, 0.0f);
        Xacc[i]  = 0.0f;
    }
}

// ---------------------------------------------------------------------------
// Final head: P = relu(X @ Wf + bf); P_ = fixed!=0 ? fixed : P; bal = inj
// ---------------------------------------------------------------------------
__global__ void final_node_kernel(const float* __restrict__ Xcur, const float* __restrict__ xnode,
                                  const float* __restrict__ Wf, const float* __restrict__ bf,
                                  float* __restrict__ Pout, float* __restrict__ Pfix,
                                  float* __restrict__ bal, float* __restrict__ scal, int N_) {
    int i = blockIdx.x * blockDim.x + threadIdx.x;
    if (i == 0) *scal = 0.0f;
    if (i < N_) {
        float acc = bf[0];
        #pragma unroll
        for (int j = 0; j < LDIM; ++j) acc += Xcur[(size_t)i * LDIM + j] * Wf[j];
        float P = fmaxf(acc, 0.0f);
        Pout[i] = P;
        float fx = xnode[(size_t)i * 4 + 3];
        Pfix[i] = (fx != 0.0f) ? fx : P;
        bal[i]  = xnode[(size_t)i * 4 + 0];   // injection
    }
}

// ---------------------------------------------------------------------------
// Flows + balance scatter
// ---------------------------------------------------------------------------
__global__ void flows_kernel(const float* __restrict__ Pfix, const int* __restrict__ srcI,
                             const int* __restrict__ dstI, const float* __restrict__ eattr,
                             float* __restrict__ flowOut, float* __restrict__ bal, int E_) {
    int e = blockIdx.x * blockDim.x + threadIdx.x;
    if (e < E_) {
        int s = srcI[e], d = dstI[e];
        float ps = Pfix[s], pd = Pfix[d];
        float dp2 = ps * ps - pd * pd;
        float kk  = eattr[(size_t)e * 4];
        float mag = sqrtf(fabsf(dp2) / kk + 1e-6f);
        float fl  = (dp2 > 0.0f) ? mag : ((dp2 < 0.0f) ? -mag : 0.0f);
        flowOut[e] = fl;
        atomicAdd(&bal[d],  fl);
        atomicAdd(&bal[s], -fl);
    }
}

// ---------------------------------------------------------------------------
// ||balance||^2 reduction, then sqrt
// ---------------------------------------------------------------------------
__global__ void reduce_kernel(const float* __restrict__ bal, float* __restrict__ scal, int N_) {
    __shared__ float sm[256];
    int tid = threadIdx.x;
    float acc = 0.0f;
    for (int i = blockIdx.x * blockDim.x + tid; i < N_; i += gridDim.x * blockDim.x) {
        float b = bal[i];
        acc += b * b;
    }
    sm[tid] = acc;
    __syncthreads();
    for (int st = 128; st > 0; st >>= 1) {
        if (tid < st) sm[tid] += sm[tid + st];
        __syncthreads();
    }
    if (tid == 0) atomicAdd(scal, sm[0]);
}

__global__ void finalize_kernel(const float* __restrict__ scal, float* __restrict__ out1) {
    if (threadIdx.x == 0 && blockIdx.x == 0) out1[0] = sqrtf(*scal);
}

// ---------------------------------------------------------------------------
// Host-side launcher
// ---------------------------------------------------------------------------
extern "C" void kernel_launch(void* const* d_in, const int* in_sizes, int n_in,
                              void* d_out, int out_size, void* d_ws, size_t ws_size,
                              hipStream_t stream) {
    const float* x     = (const float*)d_in[0];   // (N,4)
    const float* eattr = (const float*)d_in[1];   // (E,4)
    const int*   ei    = (const int*)  d_in[2];   // (2,E)
    const float* W1    = (const float*)d_in[3];   // (L,24,16)
    const float* b1    = (const float*)d_in[4];   // (L,16)
    const float* W2    = (const float*)d_in[5];   // (L,16,6)
    const float* b2    = (const float*)d_in[6];   // (L,6)
    const float* Wf    = (const float*)d_in[7];   // (6,1)
    const float* bf    = (const float*)d_in[8];   // (1,)

    const int N_ = in_sizes[0] / 4;
    const int E_ = in_sizes[1] / 4;
    const int Llayers = in_sizes[3] / (DIN * HID);

    const int* srcI = ei;
    const int* dstI = ei + E_;

    // workspace layout (floats): deg[N] | Xcur[6N] | Xacc[6N] | Pfix[N] | bal[N] | scal[1]
    float* w    = (float*)d_ws;
    float* deg  = w;
    float* Xcur = w + (size_t)N_;
    float* Xacc = w + (size_t)N_ * 7;
    float* Pfix = w + (size_t)N_ * 13;
    float* bal  = w + (size_t)N_ * 14;
    float* scal = w + (size_t)N_ * 15;
    float* out  = (float*)d_out;

    // zero deg + Xcur + Xacc (contiguous first 13N floats)
    zero_ws_kernel<<<(N_ * 13 + 255) / 256, 256, 0, stream>>>(w, N_ * 13);
    degree_kernel<<<(E_ + 255) / 256, 256, 0, stream>>>(dstI, deg, E_);

    // persistent-style launch: 2048 blocks x 8 waves -> each wave strides tiles
    const int edgeBlocks = 2048;
    for (int l = 0; l < Llayers; ++l) {
        edge_mlp_kernel<<<edgeBlocks, 256, 0, stream>>>(
            Xcur, x, eattr, srcI, dstI,
            W1 + (size_t)l * DIN * HID, b1 + (size_t)l * HID,
            W2 + (size_t)l * HID * LDIM, b2 + (size_t)l * LDIM,
            Xacc, E_);
        node_update_kernel<<<(N_ * LDIM + 255) / 256, 256, 0, stream>>>(Xacc, deg, Xcur, N_);
    }

    final_node_kernel<<<(N_ + 255) / 256, 256, 0, stream>>>(Xcur, x, Wf, bf, out, Pfix, bal, scal, N_);
    flows_kernel<<<(E_ + 255) / 256, 256, 0, stream>>>(Pfix, srcI, dstI, eattr, out + N_, bal, E_);
    reduce_kernel<<<512, 256, 0, stream>>>(bal, scal, N_);
    finalize_kernel<<<1, 64, 0, stream>>>(scal, out + (size_t)N_ + E_);
}
